// DynamicLinear_19464791785713
// MI455X (gfx1250) — matrix operand order repaired and checked by reading, not compiled, not run on hardware
//
#include <hip/hip_runtime.h>

#define L_TOK   4096
#define BATCH   16
#define C_IN    512
#define C_OUT   512
#define C_COND  512
#define K_BANK  4
#define HID     128
#define INV_TEMP (1.0f / 30.0f)

typedef __attribute__((ext_vector_type(16))) __bf16       v16bf;
typedef __attribute__((ext_vector_type(8)))  float        v8f;
typedef __attribute__((ext_vector_type(8)))  unsigned int v8u;

// result = { hiSrc[31:16], loSrc[31:16] }  (one v_perm_b32)
static __device__ __forceinline__ unsigned pack_hi16(unsigned loSrc, unsigned hiSrc) {
  return __builtin_amdgcn_perm(hiSrc, loSrc, 0x07060302u);
}

struct SplitPair { unsigned hi, lo; };

// fp32 -> (bf16 hi, bf16 lo) for two values, packed as bf16x2 dwords.
// hi = trunc16(x); lo = trunc16(x - hi)  (x - hi is exact; total err ~2^-16)
// clang lowers (x - float(hi)) to a single v_fma_mix_f32_bf16.
static __device__ __forceinline__ SplitPair split2(float x0, float x1) {
  unsigned u0 = __float_as_uint(x0), u1 = __float_as_uint(x1);
  SplitPair s;
  s.hi = pack_hi16(u0, u1);
  float h0 = __uint_as_float(u0 & 0xFFFF0000u);
  float h1 = __uint_as_float(u1 & 0xFFFF0000u);
  s.lo = pack_hi16(__float_as_uint(x0 - h0), __float_as_uint(x1 - h1));
  return s;
}

#define WMMA_BF16(A, B, C) \
  __builtin_amdgcn_wmma_f32_16x16x32_bf16(false, (A), false, (B), (short)0, (C), false, false)

// ---------------------------------------------------------------------------
// Kernel 1: conditioning MLP -> softmax attention -> aggregated bias
// ---------------------------------------------------------------------------
__global__ __launch_bounds__(256) void dl_att(
    const float* __restrict__ cond, const float* __restrict__ fc1w,
    const float* __restrict__ fc1b, const float* __restrict__ fc2w,
    const float* __restrict__ fc2b, const float* __restrict__ bias,
    float* __restrict__ att_out, float* __restrict__ bagg_out) {
  __shared__ float h[BATCH * HID];
  __shared__ float logits[BATCH * K_BANK];
  __shared__ float att_s[BATCH * K_BANK];
  const int tid = threadIdx.x;

  // h = relu(cond @ fc1_w^T + fc1_b)
  for (int t = tid; t < BATCH * HID; t += 256) {
    int b = t / HID, j = t % HID;
    float s = fc1b[j];
    const float* c = cond + b * C_COND;
    const float* w = fc1w + j * C_COND;
    for (int i = 0; i < C_COND; ++i) s += c[i] * w[i];
    h[t] = s > 0.f ? s : 0.f;
  }
  __syncthreads();

  // logits = (h @ fc2_w^T + fc2_b) / T
  if (tid < BATCH * K_BANK) {
    int b = tid / K_BANK, k = tid % K_BANK;
    float s = fc2b[k];
    const float* hv = h + b * HID;
    const float* w = fc2w + k * HID;
    for (int j = 0; j < HID; ++j) s += hv[j] * w[j];
    logits[tid] = s * INV_TEMP;
  }
  __syncthreads();

  // softmax over K
  if (tid < BATCH) {
    float m = logits[tid * K_BANK];
    for (int k = 1; k < K_BANK; ++k) m = fmaxf(m, logits[tid * K_BANK + k]);
    float e[K_BANK], sum = 0.f;
    for (int k = 0; k < K_BANK; ++k) { e[k] = __expf(logits[tid * K_BANK + k] - m); sum += e[k]; }
    float inv = 1.f / sum;
    for (int k = 0; k < K_BANK; ++k) {
      float a = e[k] * inv;
      att_s[tid * K_BANK + k] = a;
      att_out[tid * K_BANK + k] = a;
    }
  }
  __syncthreads();

  // b_agg = att @ bias
  for (int t = tid; t < BATCH * C_OUT; t += 256) {
    int b = t / C_OUT, o = t % C_OUT;
    float s = 0.f;
    for (int k = 0; k < K_BANK; ++k) s += att_s[b * K_BANK + k] * bias[k * C_OUT + o];
    bagg_out[t] = s;
  }
}

// ---------------------------------------------------------------------------
// Kernel 2: w_agg[b] = sum_k att[b,k]*weight[k], split into bf16 hi/lo planes
// ---------------------------------------------------------------------------
__global__ __launch_bounds__(256) void dl_wagg(
    const float* __restrict__ weight, const float* __restrict__ att,
    unsigned short* __restrict__ whi, unsigned short* __restrict__ wlo) {
  const int e = (blockIdx.x * 256 + threadIdx.x) * 4;   // flat over B*C_OUT*C_IN
  const int b = e >> 18;                                // 262144 elems per batch
  const int r = e & 0x3FFFF;
  const float a0 = att[b * K_BANK + 0];
  const float a1 = att[b * K_BANK + 1];
  const float a2 = att[b * K_BANK + 2];
  const float a3 = att[b * K_BANK + 3];
  const float4 w0 = *(const float4*)(weight + 0 * 262144 + r);
  const float4 w1 = *(const float4*)(weight + 1 * 262144 + r);
  const float4 w2 = *(const float4*)(weight + 2 * 262144 + r);
  const float4 w3 = *(const float4*)(weight + 3 * 262144 + r);
  float c0 = a0 * w0.x + a1 * w1.x + a2 * w2.x + a3 * w3.x;
  float c1 = a0 * w0.y + a1 * w1.y + a2 * w2.y + a3 * w3.y;
  float c2 = a0 * w0.z + a1 * w1.z + a2 * w2.z + a3 * w3.z;
  float c3 = a0 * w0.w + a1 * w1.w + a2 * w2.w + a3 * w3.w;
  SplitPair p01 = split2(c0, c1);
  SplitPair p23 = split2(c2, c3);
  *(uint2*)(whi + e) = make_uint2(p01.hi, p23.hi);
  *(uint2*)(wlo + e) = make_uint2(p01.lo, p23.lo);
}

// ---------------------------------------------------------------------------
// Kernel 3: out[l,b,:] = x[l,b,:] @ w_agg[b]^T + b_agg[b]   (bf16x3 WMMA)
// Workgroup tile 128(M) x 256(N); 8 waves arranged 4(M) x 2(N), each wave
// computing a 32(M) x 128(N) tile (2 A-frags x 8 B-frags, 48 WMMA / k-step).
// This arrangement minimizes redundant fp32->bf16x2 split work (VALU), which
// is the only non-WMMA compute in the loop; B-fragment reloads are pure L1
// traffic with a K-contiguous layout.
// ---------------------------------------------------------------------------
__global__ __launch_bounds__(256) void dl_gemm(
    const float* __restrict__ x, const unsigned short* __restrict__ whi,
    const unsigned short* __restrict__ wlo, const float* __restrict__ bagg,
    float* __restrict__ out) {
  const int tid  = threadIdx.x;
  const int wave = tid >> 5;
  const int lane = tid & 31;
  const int r    = lane & 15;   // A row / B,C column within fragment
  const int half = lane >> 4;   // lane-half selects K sub-block

  const int blk = blockIdx.x;
  const int nt = blk & 1;          // 2 N tiles of 256
  const int mt = (blk >> 1) & 31;  // 32 M tiles of 128
  const int b  = blk >> 6;         // 16 batches

  const int wm = wave & 3;         // 0..3  -> 32 rows each
  const int wn = wave >> 2;        // 0..1  -> 128 cols each
  const int l0 = mt * 128 + wm * 32;
  const int o0 = nt * 256 + wn * 128;

  // A fragment base pointers: row l0+16m+r, starting K = half*16
  const float* xrow[2];
#pragma unroll
  for (int m = 0; m < 2; ++m)
    xrow[m] = x + ((l0 + m * 16 + r) * BATCH + b) * C_IN + half * 16;

  // B fragment base pointers: col o = o0+16n+r, 16 contiguous K at half*16
  const v16bf* wh[8];
  const v16bf* wl[8];
#pragma unroll
  for (int n = 0; n < 8; ++n) {
    int base = (b * C_OUT + (o0 + n * 16 + r)) * C_IN + half * 16;
    wh[n] = (const v16bf*)(whi + base);
    wl[n] = (const v16bf*)(wlo + base);
  }

  v8f acc[2][8];
#pragma unroll
  for (int m = 0; m < 2; ++m)
#pragma unroll
    for (int n = 0; n < 8; ++n) acc[m][n] = (v8f){0.f, 0.f, 0.f, 0.f, 0.f, 0.f, 0.f, 0.f};

  for (int k = 0; k < C_IN; k += 32) {
    v16bf ahi[2], alo[2];
#pragma unroll
    for (int m = 0; m < 2; ++m) {
      const float* p = xrow[m] + k;
      float4 f0 = *(const float4*)(p);        // K = k0 .. k0+3
      float4 f1 = *(const float4*)(p + 4);    // K = k0+4 .. k0+7
      float4 f2 = *(const float4*)(p + 8);    // K = k0+8 .. k0+11
      float4 f3 = *(const float4*)(p + 12);   // K = k0+12 .. k0+15
      v8u hi, lo;
      SplitPair s;
      s = split2(f0.x, f0.y); hi[0] = s.hi; lo[0] = s.lo;
      s = split2(f0.z, f0.w); hi[1] = s.hi; lo[1] = s.lo;
      s = split2(f1.x, f1.y); hi[2] = s.hi; lo[2] = s.lo;
      s = split2(f1.z, f1.w); hi[3] = s.hi; lo[3] = s.lo;
      s = split2(f2.x, f2.y); hi[4] = s.hi; lo[4] = s.lo;
      s = split2(f2.z, f2.w); hi[5] = s.hi; lo[5] = s.lo;
      s = split2(f3.x, f3.y); hi[6] = s.hi; lo[6] = s.lo;
      s = split2(f3.z, f3.w); hi[7] = s.hi; lo[7] = s.lo;
      ahi[m] = __builtin_bit_cast(v16bf, hi);
      alo[m] = __builtin_bit_cast(v16bf, lo);
    }

    v16bf bhi[8], blo[8];
#pragma unroll
    for (int n = 0; n < 8; ++n) {
      bhi[n] = wh[n][k >> 4];
      blo[n] = wl[n][k >> 4];
    }

#pragma unroll
    for (int m = 0; m < 2; ++m)
#pragma unroll
      for (int n = 0; n < 8; ++n) {
        acc[m][n] = WMMA_BF16(ahi[m], bhi[n], acc[m][n]);  // hi*hi
        acc[m][n] = WMMA_BF16(alo[m], bhi[n], acc[m][n]);  // lo*hi
        acc[m][n] = WMMA_BF16(ahi[m], blo[n], acc[m][n]);  // hi*lo
      }
  }

  // Epilogue: C/D layout M = j + 8*half, N = r. Add b_agg, store.
#pragma unroll
  for (int n = 0; n < 8; ++n) {
    const int o = o0 + n * 16 + r;
    const float bs = bagg[b * C_OUT + o];
#pragma unroll
    for (int m = 0; m < 2; ++m) {
      const int row0 = l0 + m * 16 + half * 8;
      float* op = out + (row0 * BATCH + b) * C_OUT + o;
#pragma unroll
      for (int j = 0; j < 8; ++j)
        op[j * BATCH * C_OUT] = acc[m][n][j] + bs;
    }
  }
}

// ---------------------------------------------------------------------------
extern "C" void kernel_launch(void* const* d_in, const int* in_sizes, int n_in,
                              void* d_out, int out_size, void* d_ws, size_t ws_size,
                              hipStream_t stream) {
  (void)in_sizes; (void)n_in; (void)out_size; (void)ws_size;
  const float* x      = (const float*)d_in[0];
  const float* cond   = (const float*)d_in[1];
  const float* fc1w   = (const float*)d_in[2];
  const float* fc1b   = (const float*)d_in[3];
  const float* fc2w   = (const float*)d_in[4];
  const float* fc2b   = (const float*)d_in[5];
  const float* weight = (const float*)d_in[6];
  const float* bias   = (const float*)d_in[7];
  float* out = (float*)d_out;

  // workspace layout: [att 256B][b_agg 32KB][pad to 64KB][whi 8MB][wlo 8MB]
  float* att  = (float*)d_ws;
  float* bagg = (float*)((char*)d_ws + 256);
  unsigned short* whi = (unsigned short*)((char*)d_ws + 65536);
  unsigned short* wlo = whi + (size_t)BATCH * C_OUT * C_IN;

  dl_att<<<1, 256, 0, stream>>>(cond, fc1w, fc1b, fc2w, fc2b, bias, att, bagg);
  dl_wagg<<<(BATCH * C_OUT * C_IN) / (256 * 4), 256, 0, stream>>>(weight, att, whi, wlo);
  dl_gemm<<<BATCH * (L_TOK / 128) * (C_OUT / 256), 256, 0, stream>>>(x, whi, wlo, bagg, out);
}
